// SimpleMemoryCell_57629871177787
// MI455X (gfx1250) — matrix operand order repaired and checked by reading, not compiled, not run on hardware
//
#include <hip/hip_runtime.h>

typedef __attribute__((ext_vector_type(16))) _Float16 v16h;
typedef __attribute__((ext_vector_type(2)))  _Float16 v2h;
typedef __attribute__((ext_vector_type(8)))  float    v8f;
typedef _Float16 half_t;

#define B_DIM 256
#define T_DIM 512
#define E_DIM 256
#define U_DIM 256
#define X_W   512   // E + U
#define M_DIM 512

// ---- CDNA5 async global->LDS copy (ASYNCcnt-tracked), inline asm ----------
__device__ __forceinline__ void async_ld_b128(uint32_t lds_byte_off, const void* g) {
  asm volatile("global_load_async_to_lds_b128 %0, %1, off"
               :: "v"(lds_byte_off), "v"(g) : "memory");
}
__device__ __forceinline__ void wait_async0() {
  asm volatile("s_wait_asynccnt 0x0" ::: "memory");
}
__device__ __forceinline__ uint32_t lds_off(const void* p) {
  return (uint32_t)(uintptr_t)p;   // low 32 bits of generic ptr = LDS offset
}

// ---------------------------------------------------------------------------
// Kernel 0: W (E x U, f32) -> Wt (U x E, f16).  One-time transform so GEMM
// B-fragment k-pairs are contiguous halfs (single b32 loads, no cvt).
// ---------------------------------------------------------------------------
__global__ __launch_bounds__(256) void convert_wt(
    const float* __restrict__ Wk, const float* __restrict__ Wv,
    half_t* __restrict__ Wkt, half_t* __restrict__ Wvt)
{
  const int u = blockIdx.x;
  const int e = threadIdx.x;
  Wkt[(size_t)u * E_DIM + e] = (half_t)Wk[(size_t)e * U_DIM + u];
  Wvt[(size_t)u * E_DIM + e] = (half_t)Wv[(size_t)e * U_DIM + u];
}

// ---------------------------------------------------------------------------
// Kernel 1: mk = e @ W_key, mv = e @ W_value  (f16 WMMA, f32 accum)
// Block: 256 thr (8 waves). Tile: 32 rows x 256 cols x {mk,mv}.
// All tile staging via GLOBAL_LOAD_ASYNC_TO_LDS_B128 (ASYNCcnt).
// e streamed from HBM once; Wt slices are L2-resident re-reads.
// ---------------------------------------------------------------------------
__global__ __launch_bounds__(256) void proj_f16_wmma(
    const float* __restrict__ x, const half_t* __restrict__ Wkt,
    const half_t* __restrict__ Wvt, half_t* __restrict__ mk,
    half_t* __restrict__ mv)
{
  __shared__ float  lds_e32[32 * 36];   // e tile, fp32, stride 36 (16B rows)
  __shared__ half_t lds_wk[256 * 40];   // Wkt tile: [u][k] f16, stride 40
  __shared__ half_t lds_wv[256 * 40];

  const int tid  = threadIdx.x;
  const int wid  = tid >> 5;
  const int lane = tid & 31;
  const int nl   = lane & 15;
  const int hf   = lane >> 4;
  const int row0 = blockIdx.x * 32;
  const int rt   = wid >> 2;
  const int cg   = wid & 3;

  v8f acc[8];
  #pragma unroll
  for (int i = 0; i < 8; i++) acc[i] = v8f{0.f,0.f,0.f,0.f,0.f,0.f,0.f,0.f};

  const int arow = rt * 16 + nl;
  const uint32_t e_base  = lds_off(lds_e32);
  const uint32_t wk_base = lds_off(lds_wk);
  const uint32_t wv_base = lds_off(lds_wv);

  for (int kk = 0; kk < E_DIM; kk += 32) {
    __syncthreads();
    { // e tile: 32 rows x 32 f32 = 256 x 16B chunks, 1 per thread
      int r = tid >> 3, c4 = (tid & 7) * 4;
      async_ld_b128(e_base + (uint32_t)(r * 36 + c4) * 4,
                    x + (size_t)(row0 + r) * X_W + kk + c4);
    }
    // W tiles: 256 u-rows x 32 k f16 = 1024 x 16B chunks each, 4 per thread
    #pragma unroll
    for (int i = 0; i < 4; i++) {
      int q = tid + 256 * i;
      int row = q >> 2, part = q & 3;           // part: 16B (8 halfs) units
      uint32_t lo = (uint32_t)(row * 40 + part * 8) * 2;
      size_t   go = (size_t)row * E_DIM + kk + part * 8;
      async_ld_b128(wk_base + lo, Wkt + go);
      async_ld_b128(wv_base + lo, Wvt + go);
    }
    wait_async0();
    __syncthreads();

    // A fragment: row = arow, k0 = (v/4)*16 + hf*8 + (v%4)*2; cvt f32->f16
    v16h afrag;
    {
      uint32_t* ap = (uint32_t*)&afrag;
      #pragma unroll
      for (int v = 0; v < 8; v++) {
        int k0 = (v >> 2) * 16 + hf * 8 + (v & 3) * 2;
        float2 e2 = *(const float2*)&lds_e32[arow * 36 + k0];
        v2h hh = { (half_t)e2.x, (half_t)e2.y };
        ap[v] = __builtin_bit_cast(uint32_t, hh);
      }
    }
    const uint32_t* wkp = (const uint32_t*)lds_wk;  // 20 u32 per row
    const uint32_t* wvp = (const uint32_t*)lds_wv;
    #pragma unroll
    for (int ct = 0; ct < 4; ct++) {
      const int col = cg * 64 + ct * 16 + nl;
      v16h bk, bv;
      uint32_t* bkp = (uint32_t*)&bk;
      uint32_t* bvp = (uint32_t*)&bv;
      #pragma unroll
      for (int v = 0; v < 8; v++) {              // B frag: k0 = (v/4)*16 + hf*8 + (v%4)*2
        int w = col * 20 + (v >> 2) * 8 + hf * 4 + (v & 3);
        bkp[v] = wkp[w];
        bvp[v] = wvp[w];
      }
      acc[ct*2+0] = __builtin_amdgcn_wmma_f32_16x16x32_f16(
          false, afrag, false, bk, (short)0, acc[ct*2+0], false, false);
      acc[ct*2+1] = __builtin_amdgcn_wmma_f32_16x16x32_f16(
          false, afrag, false, bv, (short)0, acc[ct*2+1], false, false);
    }
  }
  #pragma unroll
  for (int ct = 0; ct < 4; ct++) {
    const int col = cg * 64 + ct * 16 + nl;
    #pragma unroll
    for (int r = 0; r < 8; r++) {
      size_t row = (size_t)row0 + rt * 16 + r + 8 * hf;   // D: row r + 8*half
      mk[row * U_DIM + col] = (half_t)acc[ct*2+0][r];
      mv[row * U_DIM + col] = (half_t)acc[ct*2+1][r];
    }
  }
}

// ---------------------------------------------------------------------------
// Kernel 2: strictly-causal flash attention with zero-row softmax correction.
// Block: 256 thr = 8 waves; wave w handles 16 queries [qbase + 16w, +16).
// K is DOUBLE-BUFFERED via async-to-LDS: tile kt+1's copy is issued before
// tile kt's WMMAs and awaited (s_wait_asynccnt) only after them, so the
// ASYNCcnt DMA overlaps compute at zero VGPR cost. V (needs a transpose
// through VGPRs) is single-buffered but prefetched one tile ahead.
// m init 0, l init (512 - t): the 512-t zero rows of the reference memory.
// ---------------------------------------------------------------------------
__global__ __launch_bounds__(256) void attn_flash_wmma(
    const float* __restrict__ x, const half_t* __restrict__ mk,
    const half_t* __restrict__ mv, float* __restrict__ out)
{
  __shared__ half_t lds_k[2][32 * 264];   // [buf][key][u], 528B rows
  __shared__ half_t lds_vt[256 * 34];     // [u][key], transposed
  __shared__ half_t lds_p[8][16 * 36];    // per-wave P transpose buffer

  const int b     = blockIdx.x;
  const int qbase = blockIdx.y * 128;
  const int tid   = threadIdx.x;
  const int wid   = tid >> 5;
  const int lane  = tid & 31;
  const int nl    = lane & 15;
  const int hf    = lane >> 4;
  const int qw    = qbase + wid * 16;
  const uint32_t k_base0 = lds_off(lds_k[0]);
  const uint32_t k_base1 = lds_off(lds_k[1]);

  auto issue_k = [&](int j0, int buf) {     // 16KB contiguous -> async B128
    const size_t tb = ((size_t)b * T_DIM + j0) * U_DIM;
    const uint32_t kb = buf ? k_base1 : k_base0;
    #pragma unroll
    for (int i = 0; i < 4; i++) {
      int q = tid + 256 * i;
      int row = q >> 5, part = q & 31;      // part in 16B units
      async_ld_b128(kb + (uint32_t)(row * 528 + part * 16),
                    mk + tb + (size_t)q * 8);
    }
  };
  auto stage_v = [&](int j0) {              // transposed staging through VGPRs
    const size_t tb = ((size_t)b * T_DIM + j0) * U_DIM;
    for (int i = tid; i < 32 * 128; i += 256) {
      int r = i >> 7, u2 = (i & 127) * 2;
      v2h vv = *(const v2h*)(mv + tb + (size_t)r * U_DIM + u2);
      lds_vt[(u2    ) * 34 + r] = vv[0];
      lds_vt[(u2 + 1) * 34 + r] = vv[1];
    }
  };

  // ---- Q fragments (h slice of x), f32 -> f16, kept in registers ----------
  v16h qf[8];
  {
    const size_t xbase = ((size_t)b * T_DIM + qw + nl) * X_W + E_DIM;
    #pragma unroll
    for (int uc = 0; uc < 8; uc++) {
      uint32_t* qp = (uint32_t*)&qf[uc];
      #pragma unroll
      for (int v = 0; v < 8; v++) {
        int u = uc * 32 + (v >> 2) * 16 + hf * 8 + (v & 3) * 2;
        float2 e2 = *(const float2*)(x + xbase + u);
        v2h hh = { (half_t)e2.x, (half_t)e2.y };
        qp[v] = __builtin_bit_cast(uint32_t, hh);
      }
    }
  }

  v8f acc[16];
  #pragma unroll
  for (int i = 0; i < 16; i++) acc[i] = v8f{0.f,0.f,0.f,0.f,0.f,0.f,0.f,0.f};
  float mrow[8], lrow[8];
  #pragma unroll
  for (int r = 0; r < 8; r++) {
    int t = qw + r + 8 * hf;
    mrow[r] = 0.f;                       // zero rows participate in the max
    lrow[r] = (float)(M_DIM - t);        // Z_t zero rows, each exp(0 - m)
  }

  const int nkt = qbase / 32 + 4;        // keys 0 .. qbase+127 (strictly causal)

  // ---- prologue: stage tile 0 --------------------------------------------
  issue_k(0, 0);
  stage_v(0);
  wait_async0();
  __syncthreads();

  for (int kt = 0; kt < nkt; kt++) {
    const int j0  = kt * 32;
    const int cur = kt & 1;
    const bool more = (kt + 1) < nkt;

    if (more) {
      issue_k(j0 + 32, cur ^ 1);         // overlaps all of this tile's WMMAs
      if (tid < 128) {                   // warm next V tile (128 x 128B lines)
        const size_t nb = ((size_t)b * T_DIM + j0 + 32) * U_DIM
                        + (size_t)tid * 64;
        __builtin_prefetch(mv + nb, 0, 0);
      }
    }

    if (j0 < qw + 16) {                  // wave-uniform causal skip
      // ---- S = Q @ K^T : two 16x16 tiles (keys j0..+15, j0+16..+31) -------
      v8f sa = v8f{0.f,0.f,0.f,0.f,0.f,0.f,0.f,0.f};
      v8f sb = sa;
      const uint32_t* kp = (const uint32_t*)lds_k[cur];
      #pragma unroll
      for (int uc = 0; uc < 8; uc++) {
        v16h bka, bkb;
        uint32_t* pa = (uint32_t*)&bka;
        uint32_t* pb = (uint32_t*)&bkb;
        #pragma unroll
        for (int v = 0; v < 8; v++) {
          int kb = uc * 32 + (v >> 2) * 16 + hf * 8 + (v & 3) * 2;    // u index pair (table order)
          pa[v] = kp[((nl     ) * 264 + kb) >> 1];
          pb[v] = kp[((nl + 16) * 264 + kb) >> 1];
        }
        sa = __builtin_amdgcn_wmma_f32_16x16x32_f16(false, qf[uc], false, bka,
                                                    (short)0, sa, false, false);
        sb = __builtin_amdgcn_wmma_f32_16x16x32_f16(false, qf[uc], false, bkb,
                                                    (short)0, sb, false, false);
      }

      // ---- mask + online softmax (row reductions via lane shuffles) -------
      half_t* pwp = lds_p[wid];
      #pragma unroll
      for (int r = 0; r < 8; r++) {
        const int ql = r + 8 * hf;
        const int t  = qw + ql;
        float va = (j0 + nl      < t) ? sa[r] : -1e30f;
        float vb = (j0 + 16 + nl < t) ? sb[r] : -1e30f;
        float mx = fmaxf(va, vb);
        mx = fmaxf(mx, __shfl_xor(mx, 1));
        mx = fmaxf(mx, __shfl_xor(mx, 2));
        mx = fmaxf(mx, __shfl_xor(mx, 4));
        mx = fmaxf(mx, __shfl_xor(mx, 8));
        float mn = fmaxf(mrow[r], mx);
        float sc = __expf(mrow[r] - mn);
        float pa = __expf(va - mn);
        float pb = __expf(vb - mn);
        float rs = pa + pb;
        rs += __shfl_xor(rs, 1);
        rs += __shfl_xor(rs, 2);
        rs += __shfl_xor(rs, 4);
        rs += __shfl_xor(rs, 8);
        lrow[r] = lrow[r] * sc + rs;
        mrow[r] = mn;
        #pragma unroll
        for (int ut = 0; ut < 16; ut++) acc[ut][r] *= sc;
        pwp[ql * 36 + nl]      = (half_t)pa;   // D-layout -> LDS transpose
        pwp[ql * 36 + 16 + nl] = (half_t)pb;
      }

      // ---- O += P @ V -----------------------------------------------------
      v16h pf;                              // P as A-fragment (16 x 32 keys)
      {
        uint32_t* pp = (uint32_t*)&pf;
        const uint32_t* lp = (const uint32_t*)pwp;
        #pragma unroll
        for (int v = 0; v < 8; v++) {
          int k0 = (v >> 2) * 16 + hf * 8 + (v & 3) * 2;
          pp[v] = lp[(nl * 36 + k0) >> 1];
        }
      }
      const uint32_t* vtp = (const uint32_t*)lds_vt;
      #pragma unroll
      for (int ut = 0; ut < 16; ut++) {
        v16h bv;
        uint32_t* bp = (uint32_t*)&bv;
        #pragma unroll
        for (int v = 0; v < 8; v++) {
          int kb = (v >> 2) * 16 + hf * 8 + (v & 3) * 2;         // key pair (table order)
          bp[v] = vtp[((ut * 16 + nl) * 34 + kb) >> 1];
        }
        acc[ut] = __builtin_amdgcn_wmma_f32_16x16x32_f16(false, pf, false, bv,
                                                         (short)0, acc[ut], false, false);
      }
    }

    __syncthreads();                     // all waves done with vt + k[cur]
    if (more) stage_v(j0 + 32);          // re-stage V (prefetched -> cache hits)
    wait_async0();                       // own async K(kt+1) writes landed
    __syncthreads();                     // k[nxt] + vt visible to all
  }

  // ---- out[b][t][u] = acc / l --------------------------------------------
  #pragma unroll
  for (int ut = 0; ut < 16; ut++) {
    #pragma unroll
    for (int r = 0; r < 8; r++) {
      int t = qw + r + 8 * hf;
      out[((size_t)b * T_DIM + t) * U_DIM + ut * 16 + nl] = acc[ut][r] / lrow[r];
    }
  }
}

// ---------------------------------------------------------------------------
extern "C" void kernel_launch(void* const* d_in, const int* in_sizes, int n_in,
                              void* d_out, int out_size, void* d_ws, size_t ws_size,
                              hipStream_t stream) {
  (void)in_sizes; (void)n_in; (void)out_size; (void)ws_size;
  const float* x  = (const float*)d_in[0];
  const float* Wk = (const float*)d_in[1];
  const float* Wv = (const float*)d_in[2];
  float* out = (float*)d_out;

  const size_t NKV = (size_t)B_DIM * T_DIM * U_DIM;
  half_t* mk  = (half_t*)d_ws;                 // 64 MB
  half_t* mv  = mk + NKV;                      // 64 MB
  half_t* Wkt = mv + NKV;                      // 128 KB (f16, transposed)
  half_t* Wvt = Wkt + (size_t)U_DIM * E_DIM;   // 128 KB

  convert_wt<<<dim3(U_DIM), 256, 0, stream>>>(Wk, Wv, Wkt, Wvt);
  proj_f16_wmma<<<dim3((B_DIM * T_DIM) / 32), 256, 0, stream>>>(x, Wkt, Wvt, mk, mv);
  attn_flash_wmma<<<dim3(B_DIM, T_DIM / 128), 256, 0, stream>>>(x, mk, mv, out);
}
